// GPTModel_14431090115028
// MI455X (gfx1250) — compile-verified, matrix operand order checked by reading, not run on hardware
//
#include <hip/hip_runtime.h>
#include <stdint.h>

// Problem constants (fixed by the reference)
#define BB   32
#define TT   256
#define CC   768
#define HH   8
#define HSZ  96
#define FFD  3072
#define VV   30522
#define VP   30592          // V padded to a multiple of 128 for the LM-head GEMM
#define LL   10
#define MM   (BB*TT)        // 8192 rows

typedef __attribute__((ext_vector_type(16))) __bf16 v16bf;
typedef __attribute__((ext_vector_type(8)))  float  v8f;

union FragAB {
  v16bf    bf;
  uint32_t u32[8];
  uint16_t u16[16];
};

__device__ __forceinline__ uint16_t f2bf(float f) {
  uint32_t u = __float_as_uint(f);
  uint32_t r = u + 0x7FFFu + ((u >> 16) & 1u);   // round-to-nearest-even
  return (uint16_t)(r >> 16);
}

// K-pair base index for VGPR j of a 16-bit A/B WMMA fragment (CDNA5 layout):
// lanes 0-15 (lh=0): VGPR0..3 -> K pairs 0,2,4,6 ; VGPR4..7 -> 16,18,20,22
// lanes 16-31 (lh=1): +8
__device__ __forceinline__ int kpair_of(int j, int lh) {
  return (j < 4 ? 2*j : 2*j + 8) + 8*lh;
}

__device__ __forceinline__ v8f vzero8() {
  v8f z;
#pragma unroll
  for (int e = 0; e < 8; e++) z[e] = 0.0f;
  return z;
}

// Low 32 bits of a flat shared-memory address == LDS byte offset (ISA 10.2:
// LDS aperture maps addr[31:0] to the LDS address).
__device__ __forceinline__ uint32_t lds_off(const void* p) {
  return (uint32_t)(uintptr_t)p;
}

// CDNA5 async DMA: copy 16B per lane from global memory straight into LDS,
// bypassing VGPRs. Tracked by ASYNCcnt.
__device__ __forceinline__ void async_ld_b128(uint32_t ldsByteOff, const void* gaddr) {
  asm volatile("global_load_async_to_lds_b128 %0, %1, off"
               :: "v"(ldsByteOff), "v"(gaddr) : "memory");
}
__device__ __forceinline__ void wait_async0() {
  asm volatile("s_wait_asynccnt 0x0" ::: "memory");
}

// ---------------------------------------------------------------------------
// x = tok_emb[idx] + pos_emb        (fp32, [MM, CC])
// ---------------------------------------------------------------------------
__global__ __launch_bounds__(256)
void embed_kernel(const int* __restrict__ idx, const float* __restrict__ tok,
                  const float* __restrict__ pos, float* __restrict__ x) {
  int row = blockIdx.x;               // b*TT + t
  int t   = row % TT;
  const float* te = tok + (size_t)idx[row] * CC;
  const float* pe = pos + (size_t)t * CC;
  float* xr = x + (size_t)row * CC;
  for (int c = threadIdx.x; c < CC; c += 256)
    xr[c] = te[c] + pe[c];
}

// ---------------------------------------------------------------------------
// out_bf16 = LayerNorm(x) * g + b   (one block per row)
// ---------------------------------------------------------------------------
__global__ __launch_bounds__(256)
void layernorm_kernel(const float* __restrict__ x, const float* __restrict__ g,
                      const float* __restrict__ b, uint16_t* __restrict__ out) {
  __shared__ float red[256];
  int row = blockIdx.x;
  const float* xr = x + (size_t)row * CC;

  float s = 0.0f;
  for (int c = threadIdx.x; c < CC; c += 256) s += xr[c];
  red[threadIdx.x] = s; __syncthreads();
  for (int o = 128; o > 0; o >>= 1) {
    if (threadIdx.x < o) red[threadIdx.x] += red[threadIdx.x + o];
    __syncthreads();
  }
  float mu = red[0] * (1.0f / CC);
  __syncthreads();

  float v = 0.0f;
  for (int c = threadIdx.x; c < CC; c += 256) { float d = xr[c] - mu; v += d * d; }
  red[threadIdx.x] = v; __syncthreads();
  for (int o = 128; o > 0; o >>= 1) {
    if (threadIdx.x < o) red[threadIdx.x] += red[threadIdx.x + o];
    __syncthreads();
  }
  float rstd = rsqrtf(red[0] * (1.0f / CC) + 1e-5f);

  uint16_t* orow = out + (size_t)row * CC;
  for (int c = threadIdx.x; c < CC; c += 256)
    orow[c] = f2bf((xr[c] - mu) * rstd * g[c] + b[c]);
}

// ---------------------------------------------------------------------------
// fp32 -> bf16 conversion (generic), and LM-head padded conversion
// ---------------------------------------------------------------------------
__global__ __launch_bounds__(256)
void conv_bf16_kernel(const float* __restrict__ in, uint16_t* __restrict__ out, int n) {
  int i = blockIdx.x * 256 + threadIdx.x;
  if (i < n) out[i] = f2bf(in[i]);
}

__global__ __launch_bounds__(256)
void conv_pad_kernel(const float* __restrict__ w, uint16_t* __restrict__ out) {
  size_t i = (size_t)blockIdx.x * 256 + threadIdx.x;     // over CC*VP
  if (i >= (size_t)CC * VP) return;
  int r = (int)(i / VP), c = (int)(i % VP);
  out[i] = (c < VV) ? f2bf(w[(size_t)r * VV + c]) : (uint16_t)0;
}

// ---------------------------------------------------------------------------
// Tiled bf16 WMMA GEMM: out = act(A[M,K] @ B[K,N] + bias + residual)
// Block tile 128x128, 8 waves arranged 4 (rows) x 2 (cols); each wave owns a
// 32x64 strip = 2x4 WMMA tiles. K-step 32, double-buffered LDS:
//  - A tile DMA'd global->LDS with GLOBAL_LOAD_ASYNC_TO_LDS_B128 (no VGPR
//    staging), issued for tile k+1 before the WMMA burst on tile k.
//  - B tile staged through 2 uint4 registers (needs transpose scatter),
//    loaded before the burst, committed to the other buffer after it.
// One s_wait_asynccnt + barrier per k-step.
// ---------------------------------------------------------------------------
__global__ __launch_bounds__(256)
void gemm_kernel(const uint16_t* __restrict__ A, const uint16_t* __restrict__ Bw,
                 const float* __restrict__ bias, const float* __restrict__ resid,
                 float* __restrict__ outF, uint16_t* __restrict__ outB,
                 int M, int Nreal, int K, int ldB, int ldOut, int relu) {
  __shared__ uint16_t As[2][128 * 48];   // 128 rows x 32 k (pad to 48 -> 96B stride)
  __shared__ uint16_t Bst[2][128 * 40];  // 128 cols x 32 k (transposed, pad to 40)

  int tid  = threadIdx.x;
  int wave = tid >> 5;
  int wr   = wave >> 1;               // row group 0..3  (32 rows each)
  int wc   = wave & 1;                // col group 0..1  (64 cols each)
  int lane = tid & 31;
  int lr   = lane & 15;
  int lh   = lane >> 4;
  int m0   = blockIdx.y * 128;
  int n0   = blockIdx.x * 128;

  // Per-thread staging coordinates
  // A tile: 128 rows x 32 k = 512 16B-chunks, 2 per thread (async DMA)
  int arow[2], akc[2];
#pragma unroll
  for (int j = 0; j < 2; j++) { int c = tid + 256 * j; arow[j] = c >> 2; akc[j] = c & 3; }
  // B tile: 32 k-rows x 128 n = 512 16B-chunks, 2 per thread (register staged)
  int bkr[2], bnc[2];
#pragma unroll
  for (int j = 0; j < 2; j++) { int c = tid + 256 * j; bkr[j] = c >> 4; bnc[j] = c & 15; }

  uint32_t asOff[2];
#pragma unroll
  for (int b = 0; b < 2; b++) asOff[b] = lds_off(&As[b][0]);

  v8f acc[8];
#pragma unroll
  for (int i = 0; i < 8; i++) acc[i] = vzero8();

  // ---- prologue: tile 0 into buffer 0 ----
#pragma unroll
  for (int j = 0; j < 2; j++)
    async_ld_b128(asOff[0] + (uint32_t)(arow[j] * 48 + 8 * akc[j]) * 2,
                  A + (size_t)(m0 + arow[j]) * K + 8 * akc[j]);
  uint4 rb[2];
#pragma unroll
  for (int j = 0; j < 2; j++)
    rb[j] = *(const uint4*)(Bw + (size_t)bkr[j] * ldB + n0 + 8 * bnc[j]);
#pragma unroll
  for (int j = 0; j < 2; j++) {
    const uint16_t* ds = (const uint16_t*)&rb[j];
#pragma unroll
    for (int e = 0; e < 8; e++) Bst[0][(8 * bnc[j] + e) * 40 + bkr[j]] = ds[e];
  }
  wait_async0();
  __syncthreads();

  int buf = 0;
  for (int kb = 0; kb < K; kb += 32) {
    int nb   = buf ^ 1;
    bool more = (kb + 32) < K;

    // Issue next tile's transfers before the WMMA burst (overlap)
    if (more) {
      int kbn = kb + 32;
#pragma unroll
      for (int j = 0; j < 2; j++)
        async_ld_b128(asOff[nb] + (uint32_t)(arow[j] * 48 + 8 * akc[j]) * 2,
                      A + (size_t)(m0 + arow[j]) * K + kbn + 8 * akc[j]);
#pragma unroll
      for (int j = 0; j < 2; j++)
        rb[j] = *(const uint4*)(Bw + (size_t)(kbn + bkr[j]) * ldB + n0 + 8 * bnc[j]);
    }

    // Compute: 2 A-fragments x 4 B-fragments = 8 WMMAs
    const uint16_t* Ab = As[buf];
    const uint16_t* Bb = Bst[buf];
    FragAB af0, af1;
#pragma unroll
    for (int j = 0; j < 8; j++) {
      int kp = kpair_of(j, lh);
      af0.u32[j] = *(const uint32_t*)&Ab[(32 * wr + lr) * 48 + kp];
      af1.u32[j] = *(const uint32_t*)&Ab[(32 * wr + 16 + lr) * 48 + kp];
    }
#pragma unroll
    for (int ct = 0; ct < 4; ct++) {
      FragAB bf;
#pragma unroll
      for (int j = 0; j < 8; j++)
        bf.u32[j] = *(const uint32_t*)&Bb[(64 * wc + 16 * ct + lr) * 40 + kpair_of(j, lh)];
      acc[ct]     = __builtin_amdgcn_wmma_f32_16x16x32_bf16(
          false, af0.bf, false, bf.bf, (short)0, acc[ct], false, false);
      acc[4 + ct] = __builtin_amdgcn_wmma_f32_16x16x32_bf16(
          false, af1.bf, false, bf.bf, (short)0, acc[4 + ct], false, false);
    }

    // Commit next B tile into the other buffer (safe: nobody reads it yet)
    if (more) {
#pragma unroll
      for (int j = 0; j < 2; j++) {
        const uint16_t* ds = (const uint16_t*)&rb[j];
#pragma unroll
        for (int e = 0; e < 8; e++) Bst[nb][(8 * bnc[j] + e) * 40 + bkr[j]] = ds[e];
      }
    }

    wait_async0();          // own async DMAs into As[nb] complete
    __syncthreads();        // everyone's DMAs + B commits visible
    buf = nb;
  }

  // Epilogue
#pragma unroll
  for (int rt = 0; rt < 2; rt++) {
#pragma unroll
    for (int ct = 0; ct < 4; ct++) {
#pragma unroll
      for (int j = 0; j < 8; j++) {
        int mloc = 32 * wr + 16 * rt + j + 8 * lh;
        int gm   = m0 + mloc;
        int col  = n0 + 64 * wc + 16 * ct + lr;
        if (col < Nreal) {
          float val = acc[4 * rt + ct][j];
          if (bias)  val += bias[col];
          if (resid) val += resid[(size_t)gm * ldOut + col];
          if (relu)  val = fmaxf(val, 0.0f);
          if (outF) outF[(size_t)gm * ldOut + col] = val;
          if (outB) outB[(size_t)gm * ldOut + col] = f2bf(val);
        }
      }
    }
  }
}

// ---------------------------------------------------------------------------
// Causal attention, one workgroup per (b, h, 64-query tile).
// Stage 1: S = scale * Q K^T (WMMA, masked) into 64KB LDS (fp32).
// Stage 2: row softmax; probabilities rewritten in place as bf16
//          (row sum stored in the free upper half of the fp32 row).
// Stage 3: O = P V (WMMA), divided by row sum, stored bf16.
// ---------------------------------------------------------------------------
__global__ __launch_bounds__(256)
void attn_kernel(const uint16_t* __restrict__ q, const uint16_t* __restrict__ k,
                 const uint16_t* __restrict__ v, uint16_t* __restrict__ o) {
  __shared__ float S[64 * 256];       // 64KB

  int qt = blockIdx.x & 3;            // 4 query tiles of 64 rows
  int bh = blockIdx.x >> 2;
  int b  = bh / HH, h = bh % HH;
  int hbase = h * HSZ;
  int base  = b * TT;

  int tid  = threadIdx.x;
  int wave = tid >> 5;
  int lane = tid & 31;
  int lr   = lane & 15;
  int lh   = lane >> 4;
  const float scale = 0.1020620726159658f;    // 96^-0.5

  // -------- Stage 1: scores --------
  for (int t = wave; t < 64; t += 8) {
    int rt = t >> 4, ct = t & 15;
    v8f acc = vzero8();
    int qrow = base + 64 * qt + 16 * rt + lr;
    int krow = base + 16 * ct + lr;
#pragma unroll
    for (int kb = 0; kb < HSZ; kb += 32) {
      FragAB af, bf;
#pragma unroll
      for (int j = 0; j < 8; j++) {
        int kp = kb + kpair_of(j, lh);
        af.u32[j] = *(const uint32_t*)&q[(size_t)qrow * CC + hbase + kp];
        bf.u32[j] = *(const uint32_t*)&k[(size_t)krow * CC + hbase + kp];
      }
      acc = __builtin_amdgcn_wmma_f32_16x16x32_bf16(
          false, af.bf, false, bf.bf, (short)0, acc, false, false);
    }
#pragma unroll
    for (int j = 0; j < 8; j++) {
      int m  = 16 * rt + j + 8 * lh;
      int qg = 64 * qt + m;
      int n  = 16 * ct + lr;
      S[m * 256 + n] = (n <= qg) ? acc[j] * scale : -1e30f;
    }
  }
  __syncthreads();

  // -------- Stage 2: softmax, P stored bf16 in place --------
  if (tid < 64) {
    float* Sr = &S[tid * 256];
    float mx = -1e30f;
    for (int i = 0; i < TT; i++) mx = fmaxf(mx, Sr[i]);
    uint16_t* Pr = (uint16_t*)Sr;   // bf16 row occupies first 512B of the row
    float sum = 0.0f;
    for (int i = 0; i < TT; i++) {
      float p = __expf(Sr[i] - mx);
      sum += p;
      Pr[i] = f2bf(p);              // overwrites fp32 slot i/2 (already read)
    }
    Sr[128] = sum;                  // byte offset 512: untouched by bf16 row
  }
  __syncthreads();

  // -------- Stage 3: O = P @ V --------
  for (int t = wave; t < 24; t += 8) {        // 4 row tiles x 6 col tiles
    int rt = t / 6, ct = t % 6;
    v8f acc = vzero8();
    const uint16_t* Pr = (const uint16_t*)&S[(16 * rt + lr) * 256];
    for (int kb = 0; kb < TT; kb += 32) {
      FragAB af, bf;
#pragma unroll
      for (int j = 0; j < 8; j++) {
        int kp = kb + kpair_of(j, lh);
        af.u32[j]     = *(const uint32_t*)&Pr[kp];
        bf.u16[2*j]   = v[(size_t)(base + kp)     * CC + hbase + 16 * ct + lr];
        bf.u16[2*j+1] = v[(size_t)(base + kp + 1) * CC + hbase + 16 * ct + lr];
      }
      acc = __builtin_amdgcn_wmma_f32_16x16x32_bf16(
          false, af.bf, false, bf.bf, (short)0, acc, false, false);
    }
#pragma unroll
    for (int j = 0; j < 8; j++) {
      int m   = 16 * rt + j + 8 * lh;
      float l = S[m * 256 + 128];
      int row = base + 64 * qt + m;
      o[(size_t)row * CC + hbase + 16 * ct + lr] = f2bf(acc[j] / l);
    }
  }
}

// ---------------------------------------------------------------------------
// Per-row cross-entropy:  rowloss[r] = logsumexp(logits_r) - logits_r[t_r]
// ---------------------------------------------------------------------------
__global__ __launch_bounds__(256)
void rowloss_kernel(const float* __restrict__ logits, const int* __restrict__ tgt,
                    float* __restrict__ rowloss) {
  __shared__ float red[256];
  int row = blockIdx.x;
  const float* lrow = logits + (size_t)row * VV;

  float mx = -1e30f;
  for (int i = threadIdx.x; i < VV; i += 256) mx = fmaxf(mx, lrow[i]);
  red[threadIdx.x] = mx; __syncthreads();
  for (int o = 128; o > 0; o >>= 1) {
    if (threadIdx.x < o) red[threadIdx.x] = fmaxf(red[threadIdx.x], red[threadIdx.x + o]);
    __syncthreads();
  }
  mx = red[0]; __syncthreads();

  float s = 0.0f;
  for (int i = threadIdx.x; i < VV; i += 256) s += __expf(lrow[i] - mx);
  red[threadIdx.x] = s; __syncthreads();
  for (int o = 128; o > 0; o >>= 1) {
    if (threadIdx.x < o) red[threadIdx.x] += red[threadIdx.x + o];
    __syncthreads();
  }
  if (threadIdx.x == 0)
    rowloss[row] = logf(red[0]) + mx - lrow[tgt[row]];
}

__global__ __launch_bounds__(256)
void reduceloss_kernel(const float* __restrict__ rowloss, float* __restrict__ out) {
  __shared__ float red[256];
  float s = 0.0f;
  for (int i = threadIdx.x; i < MM; i += 256) s += rowloss[i];   // fixed order
  red[threadIdx.x] = s; __syncthreads();
  for (int o = 128; o > 0; o >>= 1) {
    if (threadIdx.x < o) red[threadIdx.x] += red[threadIdx.x + o];
    __syncthreads();
  }
  if (threadIdx.x == 0) out[0] = red[0] * (1.0f / MM);
}

// ---------------------------------------------------------------------------
extern "C" void kernel_launch(void* const* d_in, const int* in_sizes, int n_in,
                              void* d_out, int out_size, void* d_ws, size_t ws_size,
                              hipStream_t stream) {
  const int*   idx  = (const int*)  d_in[0];
  const int*   tgt  = (const int*)  d_in[1];
  const float* tok  = (const float*)d_in[2];
  const float* pos  = (const float*)d_in[3];
  const float* Wq   = (const float*)d_in[4];
  const float* Wk   = (const float*)d_in[5];
  const float* Wv   = (const float*)d_in[6];
  const float* Wp   = (const float*)d_in[7];
  const float* bp   = (const float*)d_in[8];
  const float* W1   = (const float*)d_in[9];
  const float* b1   = (const float*)d_in[10];
  const float* W2   = (const float*)d_in[11];
  const float* b2   = (const float*)d_in[12];
  const float* ln1g = (const float*)d_in[13];
  const float* ln1b = (const float*)d_in[14];
  const float* ln2g = (const float*)d_in[15];
  const float* ln2b = (const float*)d_in[16];
  const float* lnfg = (const float*)d_in[17];
  const float* lnfb = (const float*)d_in[18];
  const float* Wlm  = (const float*)d_in[19];
  const float* blm  = (const float*)d_in[20];

  // Workspace carve-up (~190 MB total)
  char* p = (char*)d_ws;
  auto alloc = [&](size_t bytes) -> void* {
    void* r = (void*)p;
    p += (bytes + 255) & ~(size_t)255;
    return r;
  };
  float*    x    = (float*)   alloc((size_t)MM * CC * 4);
  uint16_t* hb   = (uint16_t*)alloc((size_t)MM * CC * 2);
  uint16_t* qb   = (uint16_t*)alloc((size_t)MM * CC * 2);
  uint16_t* kb   = (uint16_t*)alloc((size_t)MM * CC * 2);
  uint16_t* vb   = (uint16_t*)alloc((size_t)MM * CC * 2);
  uint16_t* ab   = (uint16_t*)alloc((size_t)MM * CC * 2);
  uint16_t* ffb  = (uint16_t*)alloc((size_t)MM * FFD * 2);
  uint16_t* wbuf = (uint16_t*)alloc((size_t)CC * FFD * 2);      // max layer weight
  uint16_t* wlmb = (uint16_t*)alloc((size_t)CC * VP * 2);
  float*    rls  = (float*)   alloc((size_t)MM * 4);

  float* logits = (float*)d_out;
  float* loss   = logits + (size_t)MM * VV;

  dim3 blk(256);
  dim3 gC (CC  / 128, MM / 128);   // N=768
  dim3 gFF(FFD / 128, MM / 128);   // N=3072
  dim3 gV (VP  / 128, MM / 128);   // N=30592 (padded)

  embed_kernel<<<MM, blk, 0, stream>>>(idx, tok, pos, x);

  const int nCC  = CC * CC;
  const int nCF  = CC * FFD;
  const int gcCC = (nCC + 255) / 256;
  const int gcCF = (nCF + 255) / 256;

  for (int l = 0; l < LL; l++) {
    // ---- attention block ----
    layernorm_kernel<<<MM, blk, 0, stream>>>(x, ln1g + l * CC, ln1b + l * CC, hb);

    conv_bf16_kernel<<<gcCC, blk, 0, stream>>>(Wq + (size_t)l * nCC, wbuf, nCC);
    gemm_kernel<<<gC, blk, 0, stream>>>(hb, wbuf, nullptr, nullptr, nullptr, qb,
                                        MM, CC, CC, CC, CC, 0);
    conv_bf16_kernel<<<gcCC, blk, 0, stream>>>(Wk + (size_t)l * nCC, wbuf, nCC);
    gemm_kernel<<<gC, blk, 0, stream>>>(hb, wbuf, nullptr, nullptr, nullptr, kb,
                                        MM, CC, CC, CC, CC, 0);
    conv_bf16_kernel<<<gcCC, blk, 0, stream>>>(Wv + (size_t)l * nCC, wbuf, nCC);
    gemm_kernel<<<gC, blk, 0, stream>>>(hb, wbuf, nullptr, nullptr, nullptr, vb,
                                        MM, CC, CC, CC, CC, 0);

    attn_kernel<<<BB * HH * (TT / 64), blk, 0, stream>>>(qb, kb, vb, ab);

    conv_bf16_kernel<<<gcCC, blk, 0, stream>>>(Wp + (size_t)l * nCC, wbuf, nCC);
    gemm_kernel<<<gC, blk, 0, stream>>>(ab, wbuf, bp + l * CC, x, x, nullptr,
                                        MM, CC, CC, CC, CC, 0);

    // ---- MLP block ----
    layernorm_kernel<<<MM, blk, 0, stream>>>(x, ln2g + l * CC, ln2b + l * CC, hb);

    conv_bf16_kernel<<<gcCF, blk, 0, stream>>>(W1 + (size_t)l * nCF, wbuf, nCF);
    gemm_kernel<<<gFF, blk, 0, stream>>>(hb, wbuf, b1 + l * FFD, nullptr, nullptr, ffb,
                                         MM, FFD, CC, FFD, FFD, 1 /*relu*/);
    conv_bf16_kernel<<<gcCF, blk, 0, stream>>>(W2 + (size_t)l * nCF, wbuf, nCF);
    gemm_kernel<<<gC, blk, 0, stream>>>(ffb, wbuf, b2 + l * CC, x, x, nullptr,
                                        MM, CC, FFD, CC, CC, 0);
  }

  // ---- final LN + LM head + loss ----
  layernorm_kernel<<<MM, blk, 0, stream>>>(x, lnfg, lnfb, hb);

  conv_pad_kernel<<<(int)(((size_t)CC * VP + 255) / 256), blk, 0, stream>>>(Wlm, wlmb);
  gemm_kernel<<<gV, blk, 0, stream>>>(hb, wlmb, blm, nullptr, logits, nullptr,
                                      MM, VV, CC, VP, VV, 0);

  rowloss_kernel<<<MM, blk, 0, stream>>>(logits, tgt, rls);
  reduceloss_kernel<<<1, blk, 0, stream>>>(rls, loss);
}
